// LSTMClassifier_21294447853781
// MI455X (gfx1250) — matrix operand order repaired and checked by reading, not compiled, not run on hardware
//
#include <hip/hip_runtime.h>

typedef _Float16 h16;
typedef __attribute__((ext_vector_type(16))) _Float16 v16h;
typedef __attribute__((ext_vector_type(4)))  _Float16 v4h;
typedef __attribute__((ext_vector_type(8)))  float    v8f;
typedef __attribute__((ext_vector_type(4)))  float    v4f;

#define IN_DIM 1024
#define HDIM   256
#define G4H    1024
#define BATCH  64
#define TSTEPS 512
#define MROWS  (BATCH * TSTEPS)  // 32768, row index = t*64 + b (t-major)
#define HS_STRIDE 264            // padded row stride (halves) for Hs in LDS

union VH { v16h h; v4f f[2]; };

// Build a 16-half WMMA fragment from two 16-byte runs.
__device__ __forceinline__ v16h ld_frag(const h16* p0, const h16* p1) {
  VH u;
  u.f[0] = *(const v4f*)p0;
  u.f[1] = *(const v4f*)p1;
  return u.h;
}

__device__ __forceinline__ float tanh_hw(float x) {
#if __has_builtin(__builtin_amdgcn_tanhf)
  return __builtin_amdgcn_tanhf(x);          // v_tanh_f32
#elif __has_builtin(__builtin_amdgcn_tanh_f32)
  return __builtin_amdgcn_tanh_f32(x);
#else
  float e = __expf(2.f * x);
  return 1.f - 2.f / (e + 1.f);
#endif
}
__device__ __forceinline__ float sigm_hw(float x) { return 0.5f + 0.5f * tanh_hw(0.5f * x); }

// ------------------------------------------------- f32 -> f16 (plain, weights)
__global__ void cvt_f16_kernel(const float* __restrict__ s, h16* __restrict__ d, int n) {
  int i = blockIdx.x * blockDim.x + threadIdx.x;
  if (i < n) d[i] = (h16)s[i];
}

// ------------------------- x (B,T,IN) f32 -> x16 (T,B,IN) f16 (t-major rows)
__global__ __launch_bounds__(256) void cvt_x_tr_kernel(const float* __restrict__ s, h16* __restrict__ d) {
  const int row = blockIdx.x;            // dst row = t*64 + b
  const int t = row >> 6, b = row & 63;
  const float* sp = s + ((size_t)b * TSTEPS + t) * IN_DIM;
  h16* dp = d + (size_t)row * IN_DIM;
  for (int i = threadIdx.x; i < IN_DIM; i += 256) dp[i] = (h16)sp[i];
}

// ------------------------------------------------------------------- GEMM
// XG[M, cperm] (f16) = A[M,K] (f16) * W[N,K]^T (f16) + bias[N], with the
// gate-interleaved column permutation n = g*256 + c  ->  c*4 + g  (N == 1024).
// Block tile: 128(M) x 128(N), 8 waves. 4-deep software pipeline: two disjoint
// 4-fragment B register banks so loads stay >= 4 WMMAs ahead of their use and
// the allocator cannot collapse the pipeline. Only A overreads (<=96B, padded).
__global__ __launch_bounds__(256) void gemm_bias_kernel(
    const h16* __restrict__ A, const h16* __restrict__ W,
    const float* __restrict__ bias, h16* __restrict__ C,
    int M, int N, int K)
{
  const int wave = threadIdx.x >> 5;
  const int lane = threadIdx.x & 31;
  const int r = lane & 15, s = lane >> 4;
  const int m0 = blockIdx.y * 128 + wave * 16;
  const int n0 = blockIdx.x * 128;

  v8f acc[8] = {};
  const h16* aRow = A + (size_t)(m0 + r) * K + s * 16;
  const h16* wRow = W + (size_t)(n0 + r) * K + s * 16;   // + j*16*K per tile

  v16h a = ld_frag(aRow, aRow + 8);
  v16h bb[4], bb2[4];
#pragma unroll
  for (int j = 0; j < 4; ++j) {
    const h16* bp = wRow + (size_t)j * 16 * K;
    bb[j] = ld_frag(bp, bp + 8);
  }
  for (int k0 = 0; k0 < K; k0 += 32) {
#pragma unroll
    for (int j = 0; j < 4; ++j) {                        // tiles 4..7 @ k0
      const h16* bp = wRow + (size_t)(j + 4) * 16 * K + k0;
      bb2[j] = ld_frag(bp, bp + 8);
    }
    v16h an = ld_frag(aRow + k0 + 32, aRow + k0 + 40);   // next k (overreads once)
#pragma unroll
    for (int j = 0; j < 4; ++j)                          // operands loaded last iter
      acc[j] = __builtin_amdgcn_wmma_f32_16x16x32_f16(
          false, a, false, bb[j], (short)0, acc[j], false, false);
#pragma unroll
    for (int j = 0; j < 4; ++j) {                        // tiles 0..3 @ k0+32
      const h16* bp = wRow + (size_t)j * 16 * K + k0 + 32;
      bb[j] = ld_frag(bp, bp + 8);
    }
#pragma unroll
    for (int j = 0; j < 4; ++j)
      acc[j + 4] = __builtin_amdgcn_wmma_f32_16x16x32_f16(
          false, a, false, bb2[j], (short)0, acc[j + 4], false, false);
    a = an;
  }
#pragma unroll
  for (int j = 0; j < 8; ++j) {
    const int n = n0 + j * 16 + r;
    const int nperm = (n & 255) * 4 + (n >> 8);          // gate-interleaved col
    const float bj = bias[n];
#pragma unroll
    for (int i = 0; i < 8; ++i) {
      const int m = m0 + s * 8 + i;                      // D: m = 8*s + i
      C[(size_t)m * N + nperm] = (h16)(acc[j][i] + bj);
    }
  }
}

// ------------------------------------------------------------- LSTM recurrence
// 4 WGs x 512 threads (16 waves); WG owns batch rows [16*blk, +16) -> sync-free.
// Wave w owns h-cols [16w,16w+16) and computes gate tiles n = g*256 + 16w;
// gates + cell state c stay in this wave's registers.
// Whh residency: K=0..127 as register B-fragments (128 VGPRs/wave);
//                K=128..255 as a fragment-swizzled 256KB LDS blob.
// h ping-pong (Hs0/Hs1): gemm reads cur while update writes nxt ->
// exactly ONE barrier per step. Hseq write-back = coalesced copy of h_{t-1}.
__global__ __launch_bounds__(512) void lstm_rec_kernel(
    const h16* __restrict__ XG, const h16* __restrict__ Whh,
    h16* __restrict__ Hseq)
{
  extern __shared__ __align__(16) unsigned char smem[];
  h16* Hs0  = (h16*)smem;                              // 16 x HS_STRIDE halves
  h16* Hs1  = (h16*)(smem + 16 * HS_STRIDE * 2);       // 16 x HS_STRIDE halves
  h16* WhhL = (h16*)(smem + 2 * 16 * HS_STRIDE * 2);   // 256 KB swizzled blob

  const int b0   = blockIdx.x * 16;
  const int tid  = threadIdx.x;
  const int wave = tid >> 5, lane = tid & 31;
  const int r = lane & 15, s = lane >> 4;

  // ---- preload: swizzled LDS blob for K=128..255 (16384 chunks of 16B)
  for (int c = tid; c < 16384; c += 512) {
    const int L = c & 31, half = (c >> 5) & 1, ksL = (c >> 6) & 3;
    const int g = (c >> 8) & 3, w = c >> 10;
    const int n = g * 256 + w * 16 + (L & 15);
    const int k = (ksL + 4) * 32 + (L >> 4) * 8 + half * 16;
    *(v4f*)(WhhL + (size_t)c * 8) = *(const v4f*)(Whh + (size_t)n * HDIM + k);
  }
  // ---- preload: register B-fragments for K=0..127
  v16h breg[4][4];
#pragma unroll
  for (int g = 0; g < 4; ++g)
#pragma unroll
    for (int ks = 0; ks < 4; ++ks) {
      const h16* bp = Whh + (size_t)(g * 256 + wave * 16 + r) * HDIM + ks * 32 + s * 8;
      breg[g][ks] = ld_frag(bp, bp + 16);
    }
  // ---- init h = 0 (both buffers), c = 0 (registers)
  for (int i = tid; i < 16 * HS_STRIDE; i += 512) { Hs0[i] = (h16)0.f; Hs1[i] = (h16)0.f; }
  v8f creg = {};

  // xg for t = 0 (gate-interleaved: 4 halves per cell)
  const h16* xgbase = XG + (size_t)(b0 + 8 * s) * G4H + (wave * 16 + r) * 4;
  v4h xgb[8];
#pragma unroll
  for (int i = 0; i < 8; ++i) xgb[i] = *(const v4h*)(xgbase + (size_t)i * G4H);

  const int cm = tid >> 5;          // write-back: row 0..15
  const int ck = (tid & 31) * 8;    // write-back: half offset (16B chunk)
  const int col = wave * 16 + r;
  __syncthreads();

  auto step = [&](int t, const h16* cur, h16* nxt) {
    // acc[g] initialized from xg (bias included): D = h@Whh^T + xg
    v8f acc[4];
#pragma unroll
    for (int i = 0; i < 8; ++i) {
      acc[0][i] = (float)xgb[i][0];
      acc[1][i] = (float)xgb[i][1];
      acc[2][i] = (float)xgb[i][2];
      acc[3][i] = (float)xgb[i][3];
    }
    // prefetch next step's xg (hidden behind gemm + update)
    if (t + 1 < TSTEPS) {
      const h16* p = xgbase + (size_t)(t + 1) * 64 * G4H;
#pragma unroll
      for (int i = 0; i < 8; ++i) xgb[i] = *(const v4h*)(p + (size_t)i * G4H);
    }
    // coalesced write-back of h_{t-1} (cur is stable this step)
    if (t > 0) {
      v4f hv = *(const v4f*)(cur + cm * HS_STRIDE + ck);
      *(v4f*)(Hseq + ((size_t)((t - 1) * 64 + b0 + cm)) * HDIM + ck) = hv;
    }
    // K = 0..127 from register-resident Whh
#pragma unroll
    for (int ks = 0; ks < 4; ++ks) {
      const h16* ap = cur + r * HS_STRIDE + ks * 32 + s * 8;
      v16h a = ld_frag(ap, ap + 16);
#pragma unroll
      for (int g = 0; g < 4; ++g)
        acc[g] = __builtin_amdgcn_wmma_f32_16x16x32_f16(
            false, a, false, breg[g][ks], (short)0, acc[g], false, false);
    }
    // K = 128..255 from the swizzled LDS blob
#pragma unroll
    for (int ksL = 0; ksL < 4; ++ksL) {
      const h16* ap = cur + r * HS_STRIDE + (ksL + 4) * 32 + s * 8;
      v16h a = ld_frag(ap, ap + 16);
#pragma unroll
      for (int g = 0; g < 4; ++g) {
        const h16* bp = WhhL + (size_t)(((wave * 4 + g) * 4 + ksL) * 2) * 256 + lane * 8;
        v16h b = ld_frag(bp, bp + 256);
        acc[g] = __builtin_amdgcn_wmma_f32_16x16x32_f16(
            false, a, false, b, (short)0, acc[g], false, false);
      }
    }
    // ---- activations + state update -> nxt (no barrier needed before this:
    //      all reads hit cur, all writes hit nxt)
#pragma unroll
    for (int i = 0; i < 8; ++i) {
      const float iv = sigm_hw(acc[0][i]);
      const float fv = sigm_hw(acc[1][i]);
      const float gv = tanh_hw(acc[2][i]);
      const float ov = sigm_hw(acc[3][i]);
      const float c  = fv * creg[i] + iv * gv;
      const float h  = ov * tanh_hw(c);
      creg[i] = c;
      nxt[(8 * s + i) * HS_STRIDE + col] = (h16)h;
    }
    __syncthreads();  // update(t) visible before gemm(t+1) reads this buffer
  };

  for (int t = 0; t < TSTEPS; t += 2) {
    step(t,     Hs0, Hs1);
    step(t + 1, Hs1, Hs0);
  }
  // final write-back: h after step T-1 lives in Hs0 (T even)
  v4f hv = *(const v4f*)(Hs0 + cm * HS_STRIDE + ck);
  *(v4f*)(Hseq + ((size_t)((TSTEPS - 1) * 64 + b0 + cm)) * HDIM + ck) = hv;
}

// --------------------------------------------------------------- classifier
// out[64,8] = relu(last @ W1^T + bfc1) @ W2^T + bfc2 ; last = rows (T-1)*64+b.
__global__ __launch_bounds__(256) void fc_head_kernel(
    const h16* __restrict__ Hseq,
    const float* __restrict__ W1, const float* __restrict__ bfc1,
    const float* __restrict__ W2, const float* __restrict__ bfc2,
    float* __restrict__ out)
{
  extern __shared__ __align__(16) unsigned char smem[];
  h16*   L = (h16*)smem;                      // 64x256 f16
  float* Z = (float*)(smem + 64 * 256 * 2);   // 64x128 f32
  const int tid = threadIdx.x;
  for (int i = tid; i < 64 * 256; i += 256) {
    int b = i >> 8, j = i & 255;
    L[i] = Hseq[((size_t)(TSTEPS - 1) * 64 + b) * HDIM + j];
  }
  __syncthreads();
  for (int e = tid; e < 64 * 128; e += 256) {
    int b = e >> 7, n = e & 127;
    float sacc = bfc1[n];
    for (int k = 0; k < 256; ++k) sacc += (float)L[b * 256 + k] * W1[n * 256 + k];
    Z[e] = sacc > 0.f ? sacc : 0.f;
  }
  __syncthreads();
  for (int e = tid; e < 64 * 8; e += 256) {
    int b = e >> 3, n = e & 7;
    float sacc = bfc2[n];
    for (int k = 0; k < 128; ++k) sacc += Z[b * 128 + k] * W2[n * 128 + k];
    out[e] = sacc;
  }
}

// ------------------------------------------------------------------ launcher
extern "C" void kernel_launch(void* const* d_in, const int* in_sizes, int n_in,
                              void* d_out, int out_size, void* d_ws, size_t ws_size,
                              hipStream_t stream)
{
  const float* x    = (const float*)d_in[0];
  const float* Wih0 = (const float*)d_in[1];
  const float* Whh0 = (const float*)d_in[2];
  const float* b0   = (const float*)d_in[3];
  const float* Wih1 = (const float*)d_in[4];
  const float* Whh1 = (const float*)d_in[5];
  const float* b1   = (const float*)d_in[6];
  const float* Wih2 = (const float*)d_in[7];
  const float* Whh2 = (const float*)d_in[8];
  const float* b2   = (const float*)d_in[9];
  const float* W1   = (const float*)d_in[10];
  const float* bfc1 = (const float*)d_in[11];
  const float* W2   = (const float*)d_in[12];
  const float* bfc2 = (const float*)d_in[13];
  float* out = (float*)d_out;

  unsigned char* ws = (unsigned char*)d_ws;
  size_t off = 0;
  auto alloc = [&](size_t bytes) -> void* {
    void* p = ws + off;
    off += (bytes + 255) & ~(size_t)255;
    return p;
  };
  h16* x16    = (h16*)alloc((size_t)MROWS * IN_DIM * 2);  //  64 MB, t-major
  h16* wih0_h = (h16*)alloc((size_t)G4H * IN_DIM * 2);    //   2 MB
  h16* whh0_h = (h16*)alloc((size_t)G4H * HDIM * 2);
  h16* wih1_h = (h16*)alloc((size_t)G4H * HDIM * 2);
  h16* whh1_h = (h16*)alloc((size_t)G4H * HDIM * 2);
  h16* wih2_h = (h16*)alloc((size_t)G4H * HDIM * 2);
  h16* whh2_h = (h16*)alloc((size_t)G4H * HDIM * 2);
  h16* XG     = (h16*)alloc((size_t)MROWS * G4H * 2);     //  64 MB, gate-interleaved cols
  h16* Hseq   = (h16*)alloc((size_t)MROWS * HDIM * 2 + 256); // +tail pad for GEMM A overread

  auto cvt = [&](const float* s, h16* d, int n) {
    cvt_f16_kernel<<<(n + 255) / 256, 256, 0, stream>>>(s, d, n);
  };
  cvt_x_tr_kernel<<<dim3(MROWS), dim3(256), 0, stream>>>(x, x16);
  cvt(Wih0, wih0_h, G4H * IN_DIM);
  cvt(Whh0, whh0_h, G4H * HDIM);
  cvt(Wih1, wih1_h, G4H * HDIM);
  cvt(Whh1, whh1_h, G4H * HDIM);
  cvt(Wih2, wih2_h, G4H * HDIM);
  cvt(Whh2, whh2_h, G4H * HDIM);

  const size_t recSh = (size_t)2 * 16 * HS_STRIDE * 2 + (size_t)G4H * 128 * 2; // 16896 + 262144
  const size_t fcSh  = 64 * 256 * 2 + 64 * 128 * 4;                            // 64 KB
  const dim3 gGrid(G4H / 128, MROWS / 128);                                     // 8 x 256

  // layer 0
  gemm_bias_kernel<<<gGrid, dim3(256), 0, stream>>>(x16, wih0_h, b0, XG, MROWS, G4H, IN_DIM);
  lstm_rec_kernel<<<dim3(4), dim3(512), recSh, stream>>>(XG, whh0_h, Hseq);
  // layer 1
  gemm_bias_kernel<<<gGrid, dim3(256), 0, stream>>>(Hseq, wih1_h, b1, XG, MROWS, G4H, HDIM);
  lstm_rec_kernel<<<dim3(4), dim3(512), recSh, stream>>>(XG, whh1_h, Hseq);
  // layer 2
  gemm_bias_kernel<<<gGrid, dim3(256), 0, stream>>>(Hseq, wih2_h, b2, XG, MROWS, G4H, HDIM);
  lstm_rec_kernel<<<dim3(4), dim3(512), recSh, stream>>>(XG, whh2_h, Hseq);

  fc_head_kernel<<<dim3(1), dim3(256), fcSh, stream>>>(Hseq, W1, bfc1, W2, bfc2, out);

  (void)in_sizes; (void)n_in; (void)out_size; (void)ws_size;
}
